// Attention_31997506355363
// MI455X (gfx1250) — compile-verified, matrix operand order checked by reading, not run on hardware
//
#include <hip/hip_runtime.h>

// ---------------- CDNA5 WMMA types ----------------
typedef __attribute__((ext_vector_type(16))) __bf16 v16bf;
typedef __attribute__((ext_vector_type(8)))  float  v8f;

#define B_   2
#define N_   2048
#define H_   16
#define KDIM 1024
#define WIN_ 512

// ---------------- async global->LDS staging (gfx1250 path) ----------------
#if __has_builtin(__builtin_amdgcn_global_load_async_to_lds_b128) && \
    __has_builtin(__builtin_amdgcn_s_wait_asynccnt)
#define ASYNC_LDS_OK 1
#endif

typedef __attribute__((vector_size(4 * sizeof(int)))) int v4i_;
typedef __attribute__((address_space(1))) v4i_* gvp16_t;   // global v4i*
typedef __attribute__((address_space(3))) v4i_* lvp16_t;   // LDS    v4i*

__device__ __forceinline__ void async_cp16(void* lds, const void* gl) {
#ifdef ASYNC_LDS_OK
  __builtin_amdgcn_global_load_async_to_lds_b128(
      (gvp16_t)(uintptr_t)gl, (lvp16_t)(uintptr_t)lds, 0, 0);
#else
  *(int4*)lds = *(const int4*)gl;   // synchronous fallback (still LDS-shared)
#endif
}
__device__ __forceinline__ void async_wait_all() {
#ifdef ASYNC_LDS_OK
  __builtin_amdgcn_s_wait_asynccnt(0);
#endif
}

__device__ __forceinline__ v8f wmma_bf16(v16bf a, v16bf b, v8f c) {
  // D = A(16x32 bf16) * B(32x16 bf16) + C(16x16 f32)
  return __builtin_amdgcn_wmma_f32_16x16x32_bf16(
      false, a, false, b, (short)0, c, false, false);
}

// A/B fragment loader from bf16 row-major [rows][ld]:
// lane L<16: row = row0+L,    K = k0+{0..7, 16..23}
// lane L>=16: row = row0+L-16, K = k0+{8..15, 24..31}
__device__ __forceinline__ v16bf load_frag_bf16(const __bf16* base, int ld,
                                                int row0, int k0, int lane) {
  int r  = row0 + (lane & 15);
  int kb = k0 + ((lane & 16) ? 8 : 0);
  const __bf16* p = base + (size_t)r * ld + kb;
  v16bf a;
#pragma unroll
  for (int i = 0; i < 8; ++i) { a[i] = p[i]; a[8 + i] = p[16 + i]; }
  return a;
}

// Same fragment shape but sourced from f32 row-major, converted to bf16.
__device__ __forceinline__ v16bf load_frag_f32(const float* base, int ld,
                                               int row0, int k0, int lane) {
  int r  = row0 + (lane & 15);
  int kb = k0 + ((lane & 16) ? 8 : 0);
  const float* p = base + (size_t)r * ld + kb;
  v16bf a;
#pragma unroll
  for (int i = 0; i < 8; ++i) {
    a[i]     = (__bf16)p[i];
    a[8 + i] = (__bf16)p[16 + i];
  }
  return a;
}

// row-wise reductions across the 16-lane half groups (wave32)
__device__ __forceinline__ float rowmax16(float v) {
  v = fmaxf(v, __shfl_xor(v, 1, 32));
  v = fmaxf(v, __shfl_xor(v, 2, 32));
  v = fmaxf(v, __shfl_xor(v, 4, 32));
  v = fmaxf(v, __shfl_xor(v, 8, 32));
  return v;
}
__device__ __forceinline__ float rowsum16(float v) {
  v += __shfl_xor(v, 1, 32);
  v += __shfl_xor(v, 2, 32);
  v += __shfl_xor(v, 4, 32);
  v += __shfl_xor(v, 8, 32);
  return v;
}

// ---------------- weight convert + transpose: Wt[o][i] = bf16(W[i][o]) ----
__global__ void wt_kernel(const float* __restrict__ W, __bf16* __restrict__ Wt,
                          int in_dim, int out_dim) {
  int idx = blockIdx.x * 256 + threadIdx.x;
  if (idx >= in_dim * out_dim) return;
  int o = idx / in_dim, i = idx - o * in_dim;
  Wt[idx] = (__bf16)W[(size_t)i * out_dim + o];
}

// ---------------- projection GEMM: 16x64 tile per wave --------------------
// MODE 0: out[b][h][n][HEAD_DIM]   (row-major per-head)
// MODE 1: out[b][h][HEAD_DIM][n]   (transposed, for V matrices)
template <int HEAD_DIM, int MODE>
__global__ void __launch_bounds__(128) proj_kernel(
    const float* __restrict__ A, const __bf16* __restrict__ Wt,
    __bf16* __restrict__ outp) {
  const int lane = threadIdx.x & 31;
  const int wid  = threadIdx.x >> 5;
  const int c0   = (blockIdx.x * 4 + wid) * 64;   // 64 output cols per wave
  const int r0   = blockIdx.y * 16;
  v8f acc[4] = {{}, {}, {}, {}};
  for (int k0 = 0; k0 < KDIM; k0 += 32) {
    v16bf a = load_frag_f32(A, KDIM, r0, k0, lane);   // shared by 4 WMMAs
#pragma unroll
    for (int t = 0; t < 4; ++t) {
      v16bf b = load_frag_bf16(Wt, KDIM, c0 + t * 16, k0, lane);
      acc[t] = wmma_bf16(a, b, acc[t]);
    }
  }
  const int grp = (lane & 16) ? 8 : 0;
#pragma unroll
  for (int t = 0; t < 4; ++t) {
    const int cl = c0 + t * 16 + (lane & 15);
    const int hh = cl / HEAD_DIM;                 // compile-time shift
    const int dd = cl & (HEAD_DIM - 1);
#pragma unroll
    for (int r = 0; r < 8; ++r) {
      int row = r0 + r + grp;
      int bi = row >> 11;                          // / N_
      int ni = row & (N_ - 1);
      size_t idx = (MODE == 0)
          ? (((size_t)(bi * H_ + hh) * N_ + ni) * HEAD_DIM + dd)
          : (((size_t)(bi * H_ + hh) * HEAD_DIM + dd) * N_ + ni);
      outp[idx] = (__bf16)acc[t][r];
    }
  }
}

// ---------------- stage-1 flash attention (+sink, +SiLU) ------------------
// q[bh][n][64], k[bh][n][64], vt[bh][128][n] -> hid[bh][n][128]
// 4 waves/block share K/V tiles: double-buffered async DMA into LDS.
__global__ void __launch_bounds__(128) attn1_kernel(
    const __bf16* __restrict__ qb, const __bf16* __restrict__ kb,
    const __bf16* __restrict__ vt, const float* __restrict__ sink,
    __bf16* __restrict__ hid) {
  const int tid  = threadIdx.x;
  const int lane = tid & 31;
  const int wid  = tid >> 5;
  const int bh   = blockIdx.y;
  const int head = bh & (H_ - 1);
  const int i0   = (blockIdx.x * 4 + wid) * 16;
  const __bf16* qp = qb + (size_t)bh * N_ * 64;
  const __bf16* kp = kb + (size_t)bh * N_ * 64;
  const __bf16* vp = vt + (size_t)bh * 128 * N_;

  __shared__ __bf16 Ks[2][32 * 64];    // 2 x 4KB  : K tile, row-major
  __shared__ __bf16 Vs[2][128 * 32];   // 2 x 8KB  : V tile, [d][key]
  __shared__ __bf16 Pl[4][16 * 32];    // per-wave P transpose bounce
  __bf16* pl = Pl[wid];
  const int grp  = (lane & 16) ? 8 : 0;
  const int lcol = lane & 15;

  // stage tile starting at key jt into buffer b (block-cooperative)
  auto stage = [&](int b, int jt) {
    const __bf16* kg = kp + (size_t)jt * 64;      // 4KB contiguous
#pragma unroll
    for (int i = 0; i < 2; ++i) {
      int off = (tid + i * 128) * 8;
      async_cp16(&Ks[b][off], kg + off);
    }
#pragma unroll
    for (int i = 0; i < 4; ++i) {                 // 128 d-rows x 64B
      int d   = (tid >> 2) + i * 32;
      int seg = (tid & 3) * 8;
      async_cp16(&Vs[b][d * 32 + seg], vp + (size_t)d * N_ + jt + seg);
    }
  };

  float m_i[8], l_i[8];
  v8f O[8];
  const float sk = sink[head];
#pragma unroll
  for (int r = 0; r < 8; ++r) { m_i[r] = sk; l_i[r] = 1.0f; }
#pragma unroll
  for (int t = 0; t < 8; ++t) O[t] = (v8f){};

  stage(0, 0);
  for (int jt = 0; jt < N_; jt += 32) {
    const int buf = (jt >> 5) & 1;
    async_wait_all();
    __syncthreads();                 // buffer `buf` ready; buffer buf^1 drained
    if (jt + 32 < N_) stage(buf ^ 1, jt + 32);   // DMA next tile during compute
    const __bf16* ks = Ks[buf];
    const __bf16* vs = Vs[buf];

    v8f S0 = {}, S1 = {};
#pragma unroll
    for (int kk = 0; kk < 64; kk += 32) {
      v16bf a  = load_frag_bf16(qp, 64, i0, kk, lane);   // loop-invariant, hoisted
      v16bf b0 = load_frag_bf16(ks, 64, 0, kk, lane);
      v16bf b1 = load_frag_bf16(ks, 64, 16, kk, lane);
      S0 = wmma_bf16(a, b0, S0);
      S1 = wmma_bf16(a, b1, S1);
    }
    // mask band only intersects this tile when jt in (i0-31, i0+15+WIN]
    const bool need_mask = (jt + 31 > i0) && (jt <= i0 + 15 + WIN_);
#pragma unroll
    for (int r = 0; r < 8; ++r) {
      int row = i0 + r + grp;
      int c0 = jt + lcol, c1 = c0 + 16;
      float s0 = S0[r] * 0.125f;
      float s1 = S1[r] * 0.125f;
      if (need_mask) {
        if (c0 > row && (c0 - row) <= WIN_) s0 = -3.402823466e38f;
        if (c1 > row && (c1 - row) <= WIN_) s1 = -3.402823466e38f;
      }
      float mx = rowmax16(fmaxf(s0, s1));
      float mn = fmaxf(m_i[r], mx);
      float alpha = __expf(m_i[r] - mn);
      float p0 = __expf(s0 - mn);
      float p1 = __expf(s1 - mn);
      l_i[r] = l_i[r] * alpha + rowsum16(p0 + p1);
      m_i[r] = mn;
#pragma unroll
      for (int t = 0; t < 8; ++t) O[t][r] *= alpha;
      pl[(r + grp) * 32 + lcol]      = (__bf16)p0;   // wave-private bounce
      pl[(r + grp) * 32 + lcol + 16] = (__bf16)p1;
    }
    v16bf pa = load_frag_bf16(pl, 32, 0, 0, lane);
#pragma unroll
    for (int t = 0; t < 8; ++t) {
      v16bf bv = load_frag_bf16(vs, 32, t * 16, 0, lane);
      O[t] = wmma_bf16(pa, bv, O[t]);
    }
  }
#pragma unroll
  for (int r = 0; r < 8; ++r) {
    float inv = 1.0f / l_i[r];
    int row = i0 + r + grp;
#pragma unroll
    for (int t = 0; t < 8; ++t) {
      float x = O[t][r] * inv;
      float y = x / (1.0f + __expf(-x));   // SiLU
      hid[((size_t)bh * N_ + row) * 128 + t * 16 + lcol] = (__bf16)y;
    }
  }
}

// ---------------- stage-2 flash attention (+sink), merge heads ------------
// hid[bh][n][128], k2[bh][n][128], v2t[bh][64][n] -> out2[b][n][1024]
__global__ void __launch_bounds__(128) attn2_kernel(
    const __bf16* __restrict__ qb, const __bf16* __restrict__ kb,
    const __bf16* __restrict__ vt, const float* __restrict__ sink,
    __bf16* __restrict__ out2) {
  const int tid  = threadIdx.x;
  const int lane = tid & 31;
  const int wid  = tid >> 5;
  const int bh   = blockIdx.y;
  const int head = bh & (H_ - 1);
  const int bi   = bh >> 4;
  const int i0   = (blockIdx.x * 4 + wid) * 16;
  const __bf16* qp = qb + (size_t)bh * N_ * 128;
  const __bf16* kp = kb + (size_t)bh * N_ * 128;
  const __bf16* vp = vt + (size_t)bh * 64 * N_;

  __shared__ __bf16 Ks[2][32 * 128];   // 2 x 8KB
  __shared__ __bf16 Vs[2][64 * 32];    // 2 x 4KB
  __shared__ __bf16 Pl[4][16 * 32];
  __bf16* pl = Pl[wid];
  const int grp  = (lane & 16) ? 8 : 0;
  const int lcol = lane & 15;

  auto stage = [&](int b, int jt) {
    const __bf16* kg = kp + (size_t)jt * 128;     // 8KB contiguous
#pragma unroll
    for (int i = 0; i < 4; ++i) {
      int off = (tid + i * 128) * 8;
      async_cp16(&Ks[b][off], kg + off);
    }
#pragma unroll
    for (int i = 0; i < 2; ++i) {                 // 64 d-rows x 64B
      int d   = (tid >> 2) + i * 32;
      int seg = (tid & 3) * 8;
      async_cp16(&Vs[b][d * 32 + seg], vp + (size_t)d * N_ + jt + seg);
    }
  };

  float m_i[8], l_i[8];
  v8f O[4];
  const float sk = sink[head];
#pragma unroll
  for (int r = 0; r < 8; ++r) { m_i[r] = sk; l_i[r] = 1.0f; }
#pragma unroll
  for (int t = 0; t < 4; ++t) O[t] = (v8f){};

  stage(0, 0);
  for (int jt = 0; jt < N_; jt += 32) {
    const int buf = (jt >> 5) & 1;
    async_wait_all();
    __syncthreads();
    if (jt + 32 < N_) stage(buf ^ 1, jt + 32);
    const __bf16* ks = Ks[buf];
    const __bf16* vs = Vs[buf];

    v8f S0 = {}, S1 = {};
#pragma unroll
    for (int kk = 0; kk < 128; kk += 32) {
      v16bf a  = load_frag_bf16(qp, 128, i0, kk, lane);
      v16bf b0 = load_frag_bf16(ks, 128, 0, kk, lane);
      v16bf b1 = load_frag_bf16(ks, 128, 16, kk, lane);
      S0 = wmma_bf16(a, b0, S0);
      S1 = wmma_bf16(a, b1, S1);
    }
    const bool need_mask = (jt + 31 > i0) && (jt <= i0 + 15 + WIN_);
#pragma unroll
    for (int r = 0; r < 8; ++r) {
      int row = i0 + r + grp;
      int c0 = jt + lcol, c1 = c0 + 16;
      float s0 = S0[r] * 0.125f;            // scale fixed at DIM_HEAD**-0.5
      float s1 = S1[r] * 0.125f;
      if (need_mask) {
        if (c0 > row && (c0 - row) <= WIN_) s0 = -3.402823466e38f;
        if (c1 > row && (c1 - row) <= WIN_) s1 = -3.402823466e38f;
      }
      float mx = rowmax16(fmaxf(s0, s1));
      float mn = fmaxf(m_i[r], mx);
      float alpha = __expf(m_i[r] - mn);
      float p0 = __expf(s0 - mn);
      float p1 = __expf(s1 - mn);
      l_i[r] = l_i[r] * alpha + rowsum16(p0 + p1);
      m_i[r] = mn;
#pragma unroll
      for (int t = 0; t < 4; ++t) O[t][r] *= alpha;
      pl[(r + grp) * 32 + lcol]      = (__bf16)p0;
      pl[(r + grp) * 32 + lcol + 16] = (__bf16)p1;
    }
    v16bf pa = load_frag_bf16(pl, 32, 0, 0, lane);
#pragma unroll
    for (int t = 0; t < 4; ++t) {
      v16bf bv = load_frag_bf16(vs, 32, t * 16, 0, lane);
      O[t] = wmma_bf16(pa, bv, O[t]);
    }
  }
#pragma unroll
  for (int r = 0; r < 8; ++r) {
    float inv = 1.0f / l_i[r];
    int row = i0 + r + grp;
#pragma unroll
    for (int t = 0; t < 4; ++t) {
      float x = O[t][r] * inv;
      out2[((size_t)bi * N_ + row) * 1024 + head * 64 + t * 16 + lcol] = (__bf16)x;
    }
  }
}

// ---------------- final GEMM: d_out = out2(bf16) x WoutT(bf16) -> f32 -----
__global__ void __launch_bounds__(128) final_gemm(
    const __bf16* __restrict__ A, const __bf16* __restrict__ Wt,
    float* __restrict__ outp) {
  const int lane = threadIdx.x & 31;
  const int wid  = threadIdx.x >> 5;
  const int c0   = (blockIdx.x * 4 + wid) * 64;
  const int r0   = blockIdx.y * 16;
  v8f acc[4] = {{}, {}, {}, {}};
  for (int k0 = 0; k0 < KDIM; k0 += 32) {
    v16bf a = load_frag_bf16(A, KDIM, r0, k0, lane);
#pragma unroll
    for (int t = 0; t < 4; ++t) {
      v16bf b = load_frag_bf16(Wt, KDIM, c0 + t * 16, k0, lane);
      acc[t] = wmma_bf16(a, b, acc[t]);
    }
  }
  const int grp = (lane & 16) ? 8 : 0;
#pragma unroll
  for (int t = 0; t < 4; ++t) {
    const int cl = c0 + t * 16 + (lane & 15);
#pragma unroll
    for (int r = 0; r < 8; ++r) {
      int row = r0 + r + grp;
      outp[(size_t)row * 1024 + cl] = acc[t][r];
    }
  }
}

// --------------------------------------------------------------------------
extern "C" void kernel_launch(void* const* d_in, const int* in_sizes, int n_in,
                              void* d_out, int out_size, void* d_ws, size_t ws_size,
                              hipStream_t stream) {
  const float* Xq   = (const float*)d_in[0];
  const float* Xkv  = (const float*)d_in[1];
  const float* Wq   = (const float*)d_in[2];
  const float* Wk1  = (const float*)d_in[3];
  const float* Wv1  = (const float*)d_in[4];
  const float* Wk2  = (const float*)d_in[5];
  const float* Wv2  = (const float*)d_in[6];
  const float* Wout = (const float*)d_in[7];
  const float* sink = (const float*)d_in[8];
  float* out = (float*)d_out;

  char* ws = (char*)d_ws;
  size_t off = 0;
  auto alloc = [&](size_t bytes) {
    char* p = ws + off;
    off += (bytes + 255) & ~(size_t)255;
    return p;
  };
  __bf16* wq_t   = (__bf16*)alloc((size_t)1024 * 1024 * 2);
  __bf16* wk1_t  = (__bf16*)alloc((size_t)1024 * 1024 * 2);
  __bf16* wv1_t  = (__bf16*)alloc((size_t)1024 * 2048 * 2);
  __bf16* wk2_t  = (__bf16*)alloc((size_t)1024 * 2048 * 2);
  __bf16* wv2_t  = (__bf16*)alloc((size_t)1024 * 1024 * 2);
  __bf16* wout_t = (__bf16*)alloc((size_t)1024 * 1024 * 2);
  __bf16* q_bf   = (__bf16*)alloc((size_t)B_ * H_ * N_ * 64 * 2);
  __bf16* k1_bf  = (__bf16*)alloc((size_t)B_ * H_ * N_ * 64 * 2);
  __bf16* v1t    = (__bf16*)alloc((size_t)B_ * H_ * 128 * N_ * 2);
  __bf16* k2_bf  = (__bf16*)alloc((size_t)B_ * H_ * N_ * 128 * 2);
  __bf16* v2t    = (__bf16*)alloc((size_t)B_ * H_ * 64 * N_ * 2);
  __bf16* hid    = (__bf16*)alloc((size_t)B_ * H_ * N_ * 128 * 2);
  __bf16* out2   = (__bf16*)alloc((size_t)B_ * N_ * 1024 * 2);
  (void)ws_size; (void)in_sizes; (void)n_in; (void)out_size;

  // 1) weight convert/transpose to bf16 [out][in]
  auto wt = [&](const float* W, __bf16* Wt, int in_dim, int out_dim) {
    int total = in_dim * out_dim;
    wt_kernel<<<dim3((total + 255) / 256), dim3(256), 0, stream>>>(W, Wt, in_dim, out_dim);
  };
  wt(Wq,   wq_t,   1024, 1024);
  wt(Wk1,  wk1_t,  1024, 1024);
  wt(Wv1,  wv1_t,  1024, 2048);
  wt(Wk2,  wk2_t,  1024, 2048);
  wt(Wv2,  wv2_t,  1024, 1024);
  wt(Wout, wout_t, 1024, 1024);

  // 2) projections: rows = B_*N_ = 4096 -> 256 row tiles; 256 cols per block
  proj_kernel<64, 0><<<dim3(4, 256), dim3(128), 0, stream>>>(Xq,  wq_t,  q_bf);
  proj_kernel<64, 0><<<dim3(4, 256), dim3(128), 0, stream>>>(Xkv, wk1_t, k1_bf);
  proj_kernel<128, 1><<<dim3(8, 256), dim3(128), 0, stream>>>(Xkv, wv1_t, v1t);
  proj_kernel<128, 0><<<dim3(8, 256), dim3(128), 0, stream>>>(Xkv, wk2_t, k2_bf);
  proj_kernel<64, 1><<<dim3(4, 256), dim3(128), 0, stream>>>(Xkv, wv2_t, v2t);

  // 3) attention stages: grid = (128 row tiles / 4 waves, B_*H_)
  attn1_kernel<<<dim3(32, 32), dim3(128), 0, stream>>>(q_bf, k1_bf, v1t, sink, hid);
  attn2_kernel<<<dim3(32, 32), dim3(128), 0, stream>>>(hid, k2_bf, v2t, sink, out2);

  // 4) output projection -> f32
  final_gemm<<<dim3(4, 256), dim3(128), 0, stream>>>(out2, wout_t, out);
}
